// HyperEdgeConv_36807869726832
// MI455X (gfx1250) — matrix operand order repaired and checked by reading, not compiled, not run on hardware
//
#include <hip/hip_runtime.h>

// ---------------------------------------------------------------------------
// 2-layer GCN forward for MI455X (gfx1250, wave32, WMMA, TDM).
//   h1   = x @ W1                      (bf16 WMMA, f32 accum; A/B pre-swizzled)
//   a1   = D^-1/2 (A+I) D^-1/2 h1 + b1 (atomic scatter over edges)
//   hd   = dropout(relu(a1))           (fused into bf16 A-fragment repack)
//   h2   = hd @ W2                     (bf16 WMMA)
//   out  = D^-1/2 (A+I) D^-1/2 h2 + b2
// ---------------------------------------------------------------------------

typedef __attribute__((ext_vector_type(16))) __bf16        v16bf;
typedef __attribute__((ext_vector_type(8)))  float         v8f;
typedef __attribute__((ext_vector_type(4)))  unsigned int  u32x4;
typedef __attribute__((ext_vector_type(8)))  int           i32x8;
typedef __attribute__((ext_vector_type(4)))  int           i32x4;

union BfVec {
    v16bf        v;
    unsigned int u[8];
    u32x4        q[2];
};

__device__ __forceinline__ unsigned int pack_bf16(float lo, float hi) {
    unsigned int ulo = __float_as_uint(lo);
    unsigned int uhi = __float_as_uint(hi);
    // round-to-nearest-even f32 -> bf16
    ulo = (ulo + 0x7FFFu + ((ulo >> 16) & 1u)) >> 16;
    uhi = (uhi + 0x7FFFu + ((uhi >> 16) & 1u)) >> 16;
    return (ulo & 0xFFFFu) | (uhi << 16);
}

// ---------------------------------------------------------------------------
// TDM: DMA a linear run of n_dwords from global memory into LDS.
// D# per CDNA5 ISA (2D tensor, single-row tile).
// Mandatory: if the builtin is absent or the arity is wrong this fails to
// compile, which is the signal we want in a compile-only loop.
// ---------------------------------------------------------------------------
__device__ __forceinline__ void tdm_load_to_lds(const void* gptr, unsigned int lds_off,
                                                unsigned int n_dwords) {
    unsigned long long ga = (unsigned long long)(uintptr_t)gptr;
    u32x4 g0;
    g0[0] = 1u;                                   // count=1 valid user descriptor
    g0[1] = lds_off;                              // lds_addr (bytes)
    g0[2] = (unsigned int)ga;                     // global_addr[31:0]   (bits 95:64)
    g0[3] = (unsigned int)((ga >> 32) & 0x01FFFFFFull) | 0x80000000u; // addr[56:32], type=2

    i32x8 g1;
    g1[0] = (int)(2u << 16);                      // workgroup_mask=0, data_size=4B
    g1[1] = (int)((n_dwords & 0xFFFFu) << 16);    // tensor_dim0[15:0] @ [63:48]
    g1[2] = (int)((n_dwords >> 16) | (1u << 16)); // tensor_dim0[31:16], tensor_dim1=1
    g1[3] = (int)((n_dwords & 0xFFFFu) << 16);    // tensor_dim1 hi=0, tile_dim0=n_dwords
    g1[4] = (int)1;                               // tile_dim1=1, tile_dim2=0
    g1[5] = (int)n_dwords;                        // tensor_dim0_stride[31:0]
    g1[6] = 0;                                    // stride0 hi, stride1 lo
    g1[7] = 0;
    i32x4 z4 = {0, 0, 0, 0};
#if __clang_major__ >= 23
    i32x8 z8 = {0, 0, 0, 0, 0, 0, 0, 0};
    __builtin_amdgcn_tensor_load_to_lds(g0, g1, z4, z4, z8, 0);
#else
    __builtin_amdgcn_tensor_load_to_lds(g0, g1, z4, z4, 0);
#endif
}

// ---------------------------------------------------------------------------
// Pre-swizzle a KxN f32 weight matrix into WMMA B-fragment layout (bf16 pairs).
// B tile is 32x16 (KxN): lane l -> n = l&15, kb = (l>>4)*8; dwords 0..3 pack
// K = kb+2v,+1; dwords 4..7 pack K = kb+16+2v,+1.  frag index = ((kt*NT + nt)*32 + lane)*8 + v
// ---------------------------------------------------------------------------
__global__ void prep_wfrag(const float* __restrict__ W,
                           unsigned int* __restrict__ frag,
                           int KT, int NT, int Ncols) {
    int idx = blockIdx.x * blockDim.x + threadIdx.x;
    int total = KT * NT * 256;
    if (idx >= total) return;
    int v    = idx & 7;
    int lane = (idx >> 3) & 31;
    int t    = idx >> 8;          // kt*NT + nt
    int nt   = t % NT;
    int kt   = t / NT;
    int n    = nt * 16 + (lane & 15);
    int k0   = afrag_k(kt, lane, v);
    frag[idx] = pack_bf16(W[k0 * Ncols + n], W[(k0 + 1) * Ncols + n]);
}

// ---------------------------------------------------------------------------
// Pre-pack an [Mtiles*16 x KT*32] f32 matrix into the WMMA 16-bit A-fragment
// layout: lane l -> m = l&15, kb = (l>>4)*8; dwords 0..3 pack K = kb+2v,+1;
// dwords 4..7 pack K = kb+16+2v,+1.  frag index = ((mt*KT + kt)*32+lane)*8+v
// ---------------------------------------------------------------------------
__device__ __forceinline__ int afrag_k(int kt, int lane, int v) {
    return kt * 32 + (lane >> 4) * 8 + ((v < 4) ? 2 * v : 16 + 2 * (v - 4));
}

__global__ void prep_afrag(const float* __restrict__ A,
                           unsigned int* __restrict__ frag,
                           int Mtiles, int KT) {
    int idx = blockIdx.x * blockDim.x + threadIdx.x;
    int total = Mtiles * KT * 256;
    if (idx >= total) return;
    int v    = idx & 7;
    int lane = (idx >> 3) & 31;
    int t    = idx >> 8;          // mt*KT + kt
    int kt   = t % KT;
    int mt   = t / KT;
    int row  = mt * 16 + (lane & 15);
    const float* p = A + (size_t)row * (KT * 32) + afrag_k(kt, lane, v);
    frag[idx] = pack_bf16(p[0], p[1]);
}

// ---------------------------------------------------------------------------
// Fused ReLU + dropout(p=0.5, scale x2) + bf16 A-fragment repack (K=128, KT=4).
// Deterministic hash mask keyed on the element's linear index.
// ---------------------------------------------------------------------------
__device__ __forceinline__ float relu_drop(float v, unsigned int idx) {
    v = v > 0.0f ? v : 0.0f;
    unsigned int x = idx * 2654435761u ^ 42u;
    x ^= x >> 16; x *= 0x7feb352du;
    x ^= x >> 15; x *= 0x846ca68bu;
    x ^= x >> 16;
    return (x & 1u) ? v * 2.0f : 0.0f;
}

__global__ void relu_dropout_frag(const float* __restrict__ a1,
                                  unsigned int* __restrict__ frag,
                                  int Mtiles) {
    const int KT = 4;                 // K = 128
    int idx = blockIdx.x * blockDim.x + threadIdx.x;
    int total = Mtiles * KT * 256;
    if (idx >= total) return;
    int v    = idx & 7;
    int lane = (idx >> 3) & 31;
    int t    = idx >> 8;
    int kt   = t % KT;
    int mt   = t / KT;
    int row  = mt * 16 + (lane & 15);
    unsigned int base = (unsigned int)(row * 128 + afrag_k(kt, lane, v));
    float x0 = relu_drop(a1[base],     base);
    float x1 = relu_drop(a1[base + 1], base + 1);
    frag[idx] = pack_bf16(x0, x1);
}

// ---------------------------------------------------------------------------
// Degree (in-degree + self loop) and D^-1/2
// ---------------------------------------------------------------------------
__global__ void deg_init(unsigned int* __restrict__ cnt, int N) {
    int i = blockIdx.x * blockDim.x + threadIdx.x;
    if (i < N) cnt[i] = 1u;   // self loop
}

__global__ void deg_count(const long long* __restrict__ ei,
                          unsigned int* __restrict__ cnt, int E) {
    int e = blockIdx.x * blockDim.x + threadIdx.x;
    if (e >= E) return;
    atomicAdd(&cnt[(int)ei[E + e]], 1u);   // dst = edge_index[1][e]
}

__global__ void deg_rsqrt(const unsigned int* __restrict__ cnt,
                          float* __restrict__ dinv, int N) {
    int i = blockIdx.x * blockDim.x + threadIdx.x;
    if (i >= N) return;
    unsigned int d = cnt[i];
    dinv[i] = (d > 0u) ? rsqrtf((float)d) : 0.0f;
}

// ---------------------------------------------------------------------------
// GEMM: Out[Mtiles*16 x NTILES*16] = A * B, both operands pre-swizzled bf16
// fragments.  8 waves per block, one 16-row M-tile per wave.  All B fragments
// are staged once per block into LDS by the TDM, then read via ds_load_b128.
// Tail waves (tile >= Mtiles) drop out wave-uniformly after the barrier so
// EXEC stays all-ones around every WMMA.
// ---------------------------------------------------------------------------
template <int KTILES, int NTILES>
__global__ __launch_bounds__(256) void gemm_wmma(
        const unsigned int* __restrict__ Afrag,
        const unsigned int* __restrict__ Bfrag,
        float* __restrict__ Out,
        int Mtiles) {
    constexpr int NWORDS = KTILES * NTILES * 256;
    __shared__ unsigned int lds_b[NWORDS];

    if (threadIdx.x < 32) {   // wave 0 issues the tensor DMA (EXEC ignored by TDM)
        tdm_load_to_lds(Bfrag, (unsigned int)(uintptr_t)lds_b, NWORDS);
        __builtin_amdgcn_s_wait_tensorcnt(0);
    }
    __syncthreads();

    const int lane = threadIdx.x & 31;
    const int wave = threadIdx.x >> 5;
    const int tile = blockIdx.x * 8 + wave;
    if (tile >= Mtiles) return;   // wave-uniform; no barriers beyond this point

    const int half = lane >> 4;
    const int m    = lane & 15;
    const int Nc   = NTILES * 16;

    v8f acc[NTILES] = {};

#pragma unroll
    for (int kt = 0; kt < KTILES; ++kt) {
        BfVec af;
        const u32x4* ap =
            (const u32x4*)(Afrag + (((size_t)tile * KTILES + kt) << 8) + (lane << 3));
        af.q[0] = ap[0];
        af.q[1] = ap[1];
#pragma unroll
        for (int nt = 0; nt < NTILES; ++nt) {
            BfVec bf;
            const u32x4* bp =
                (const u32x4*)(lds_b + (((kt * NTILES + nt) << 8) + (lane << 3)));
            bf.q[0] = bp[0];
            bf.q[1] = bp[1];
            acc[nt] = __builtin_amdgcn_wmma_f32_16x16x32_bf16(
                false, af.v, false, bf.v, (short)0, acc[nt], false, false);
        }
    }

    const int row0 = tile * 16;
#pragma unroll
    for (int nt = 0; nt < NTILES; ++nt) {
#pragma unroll
        for (int v = 0; v < 8; ++v) {
            Out[(size_t)(row0 + half * 8 + v) * Nc + nt * 16 + m] = acc[nt][v];
        }
    }
}

// ---------------------------------------------------------------------------
// Aggregation: self-loop/bias init + per-edge atomic scatter-add
// ---------------------------------------------------------------------------
__global__ void agg_self_init(const float* __restrict__ h,
                              const float* __restrict__ dinv,
                              const float* __restrict__ bias,
                              float* __restrict__ out,
                              int total, int C) {
    int idx = blockIdx.x * blockDim.x + threadIdx.x;
    if (idx >= total) return;
    int i = idx / C;
    int c = idx - i * C;
    float di = dinv[i];
    out[idx] = di * di * h[idx] + bias[c];
}

__global__ void agg_edges(const long long* __restrict__ ei,
                          const float* __restrict__ dinv,
                          const float* __restrict__ h,
                          float* __restrict__ out,
                          int E, int C) {
    int e = blockIdx.x;            // one block per edge
    int c = threadIdx.x;           // one thread per channel (C = blockDim.x)
    int src = (int)ei[e];
    int dst = (int)ei[E + e];
    float nrm = dinv[src] * dinv[dst];
    __hip_atomic_fetch_add(&out[(size_t)dst * C + c], nrm * h[(size_t)src * C + c],
                           __ATOMIC_RELAXED, __HIP_MEMORY_SCOPE_AGENT);
}

// ---------------------------------------------------------------------------
extern "C" void kernel_launch(void* const* d_in, const int* in_sizes, int n_in,
                              void* d_out, int out_size, void* d_ws, size_t ws_size,
                              hipStream_t stream) {
    (void)in_sizes; (void)n_in; (void)out_size; (void)ws_size;

    const float*     x   = (const float*)d_in[0];
    const long long* ei  = (const long long*)d_in[1];   // int64 [2, E]
    const float*     W1  = (const float*)d_in[2];       // [256,128]
    const float*     b1  = (const float*)d_in[3];       // [128]
    const float*     W2  = (const float*)d_in[4];       // [128,64]
    const float*     b2  = (const float*)d_in[5];       // [64]
    float*           out = (float*)d_out;               // [50000,64]

    const int N = 50000, E = 800000, H = 128, C = 64;
    const int MT = N / 16;        // 3125 M-tiles (exact)

    char*  ws  = (char*)d_ws;
    size_t off = 0;
    auto carve = [&](size_t bytes) -> void* {
        void* p = ws + off;
        off += (bytes + 255) & ~(size_t)255;
        return p;
    };

    unsigned int* cnt    = (unsigned int*)carve((size_t)N * 4);
    float*        dinv   = (float*)carve((size_t)N * 4);
    unsigned int* w1frag = (unsigned int*)carve((size_t)8 * 8 * 256 * 4);   // 64 KB
    unsigned int* w2frag = (unsigned int*)carve((size_t)4 * 4 * 256 * 4);   // 16 KB
    float*        h1     = (float*)carve((size_t)N * H * 4);                // 25.6 MB
    float*        buf1   = (float*)carve((size_t)N * H * 4);                // 25.6 MB (later: h2)
    unsigned int* afragA = (unsigned int*)carve((size_t)MT * 8 * 256 * 4);  // 25.6 MB (later: afrag2)
    // aliases (lifetimes are disjoint in stream order):
    unsigned int* afrag2 = afragA;                  // layer-2 A fragments (12.8 MB)
    float*        h2     = buf1;                    // layer-2 GEMM output (12.8 MB)

    // weight pre-swizzle into WMMA B-fragment layout
    prep_wfrag<<<(8 * 8 * 256 + 255) / 256, 256, 0, stream>>>(W1, w1frag, 8, 8, H);
    prep_wfrag<<<(4 * 4 * 256 + 255) / 256, 256, 0, stream>>>(W2, w2frag, 4, 4, C);

    // x -> bf16 A fragments (KT = 8)
    prep_afrag<<<(MT * 8 * 256 + 255) / 256, 256, 0, stream>>>(x, afragA, MT, 8);

    // degree + D^-1/2
    deg_init<<<(N + 255) / 256, 256, 0, stream>>>(cnt, N);
    deg_count<<<(E + 255) / 256, 256, 0, stream>>>(ei, cnt, E);
    deg_rsqrt<<<(N + 255) / 256, 256, 0, stream>>>(cnt, dinv, N);

    // layer 1: transform (WMMA) then propagate
    gemm_wmma<8, 8><<<(MT + 7) / 8, 256, 0, stream>>>(afragA, w1frag, h1, MT);
    agg_self_init<<<(N * H + 255) / 256, 256, 0, stream>>>(h1, dinv, b1, buf1, N * H, H);
    agg_edges<<<E, H, 0, stream>>>(ei, dinv, h1, buf1, E, H);

    // relu + dropout fused into the layer-2 A-fragment repack (afragA is dead now)
    relu_dropout_frag<<<(MT * 4 * 256 + 255) / 256, 256, 0, stream>>>(buf1, afrag2, MT);

    // layer 2 (buf1 is dead after the repack; reuse it as h2)
    gemm_wmma<4, 4><<<(MT + 7) / 8, 256, 0, stream>>>(afrag2, w2frag, h2, MT);
    agg_self_init<<<(N * C + 255) / 256, 256, 0, stream>>>(h2, dinv, b2, out, N * C, C);
    agg_edges<<<E, C, 0, stream>>>(ei, dinv, h2, out, E, C);
}
